// GraphConcatAttention_19421842112974
// MI455X (gfx1250) — matrix operand order repaired and checked by reading, not compiled, not run on hardware
//
#include <hip/hip_runtime.h>
#include <hip/hip_bf16.h>
#include <stdint.h>

#define K_NBR 16
#define DIM   128
#define NHEAD 8
#define NEG_SLOPE 0.01f

#define WAVES 4
#define THREADS (WAVES * 32)
#define ITEMS_PER_WAVE 16
#define ITEMS_PER_BLOCK (WAVES * ITEMS_PER_WAVE)  // 64

typedef __attribute__((ext_vector_type(16))) __bf16 v16bf;
typedef __attribute__((ext_vector_type(8)))  __bf16 v8bf;
typedef __attribute__((ext_vector_type(8)))  float  v8f;
typedef __attribute__((ext_vector_type(4)))  int    v4i;

union ABFrag { v16bf v; v8bf h[2]; };

#if defined(__HIP_DEVICE_COMPILE__) && \
    __has_builtin(__builtin_amdgcn_global_load_async_to_lds_b128) && \
    __has_builtin(__builtin_amdgcn_s_wait_asynccnt)
#define HAVE_ASYNC_LDS 1
#else
#define HAVE_ASYNC_LDS 0
#endif

__device__ __forceinline__ unsigned bfbits(float f) {
  unsigned u = __builtin_bit_cast(unsigned, f);
  u += 0x7FFFu + ((u >> 16) & 1u);   // round-to-nearest-even
  return u >> 16;
}
__device__ __forceinline__ __bf16 f2bf(float f) {
  unsigned short h = (unsigned short)bfbits(f);
  return __builtin_bit_cast(__bf16, h);
}

// Convert 8 consecutive f32 in LDS to a v8bf half-fragment (in-register).
__device__ __forceinline__ v8bf cvt8(const float* p) {
  float4 a = *(const float4*)p;
  float4 b = *(const float4*)(p + 4);
  v8bf r;
  r[0] = f2bf(a.x); r[1] = f2bf(a.y); r[2] = f2bf(a.z); r[3] = f2bf(a.w);
  r[4] = f2bf(b.x); r[5] = f2bf(b.y); r[6] = f2bf(b.z); r[7] = f2bf(b.w);
  return r;
}

__device__ __forceinline__ v8f wmma_bf16(const ABFrag& a, const ABFrag& b, v8f c) {
  return __builtin_amdgcn_wmma_f32_16x16x32_bf16(
      /*neg_a=*/false, a.v, /*neg_b=*/false, b.v,
      /*c_mod=*/(short)0, c, /*reuse_a=*/false, /*reuse_b=*/false);
}

#if HAVE_ASYNC_LDS
// Direct global->LDS async copy (tracked by ASYNCcnt), 16B per lane.
// Builtin signature (from clang diagnostic): (v4i AS1*, v4i AS3*, i32 offset, i32 cpol).
// Low 32 bits of a generic LDS address are the LDS byte address (ISA 10.2).
typedef __attribute__((address_space(1))) v4i* gbl_v4i_ptr;
typedef __attribute__((address_space(3))) v4i* lds_v4i_ptr;
__device__ __forceinline__ void async_copy_b128(const void* g, void* lds) {
  gbl_v4i_ptr gp = (gbl_v4i_ptr)(uintptr_t)g;
  lds_v4i_ptr lp = (lds_v4i_ptr)(unsigned)(uintptr_t)lds;
  __builtin_amdgcn_global_load_async_to_lds_b128(gp, lp, 0, 0);
}
#endif

struct __align__(16) LdsLayout {
  __bf16 Wemb[DIM * DIM];                   // [n][d] bf16 of W_emb (row-major)
  __bf16 Wfin[DIM * DIM];                   // [o][d] bf16 of W_fin
  __bf16 Wsc [16 * DIM];                    // rows 0..7 = W_nbr[h], rows 8..15 = W_self[h]
  float  xs  [WAVES][K_NBR * DIM];          // f32 staging: x[n] first, then xe
  __bf16 xebf[WAVES][K_NBR * DIM];          // bf16 of xe (score WMMA A operand)
  float  sc  [WAVES][K_NBR * 16];           // raw score tile (16 cols: nbr|self)
  float  pp  [WAVES][K_NBR * NHEAD];        // softmax probs
  __bf16 row [WAVES][ITEMS_PER_WAVE * DIM]; // per-item attention output rows (bf16)
};

__global__ __launch_bounds__(THREADS)
void gca_kernel(const float* __restrict__ x,
                const unsigned char* __restrict__ mask,
                const float* __restrict__ Wemb,
                const float* __restrict__ Wscore,   // H x 2D
                const float* __restrict__ Wfin,
                float* __restrict__ out, int N) {
  __shared__ LdsLayout L;
  const int tid  = threadIdx.x;
  const int lane = tid & 31;
  const int w    = tid >> 5;
  const int half = lane >> 4;       // 0: lanes 0-15, 1: lanes 16-31
  const int l16  = lane & 15;

  // ---- preload weights to LDS as bf16 (block-cooperative) ----
  for (int i = tid; i < DIM * DIM; i += THREADS) {
    L.Wemb[i] = f2bf(Wemb[i]);
    L.Wfin[i] = f2bf(Wfin[i]);
  }
  for (int i = tid; i < 16 * DIM; i += THREADS) {
    int r = i >> 7, d = i & (DIM - 1);
    float v = (r < NHEAD) ? Wscore[r * (2 * DIM) + DIM + d]      // W_nbr
                          : Wscore[(r - NHEAD) * (2 * DIM) + d]; // W_self
    L.Wsc[i] = f2bf(v);
  }
  __syncthreads();

  const int waveBase = blockIdx.x * ITEMS_PER_BLOCK + w * ITEMS_PER_WAVE;

  for (int i = 0; i < ITEMS_PER_WAVE; ++i) {
    const int n = waveBase + i;
    const bool valid = n < N;      // wave-uniform
    if (valid && (n + 1) < N)
      __builtin_prefetch(x + (size_t)(n + 1) * K_NBR * DIM, 0, 1); // global_prefetch_b8

    // ---- stage x[n] (16x128 f32 = 8KB) into LDS ----
    if (valid) {
#if HAVE_ASYNC_LDS
      const char* gbase = (const char*)(x + (size_t)n * K_NBR * DIM);
      char*       lbase = (char*)&L.xs[w][0];
#pragma unroll
      for (int it = 0; it < 16; ++it) {
        const int off = (lane + it * 32) * 16;     // 512B per issue across the wave
        async_copy_b128(gbase + off, lbase + off);
      }
      __builtin_amdgcn_s_wait_asynccnt(0);
#else
      const float4* xf4 = (const float4*)(x + (size_t)n * K_NBR * DIM);
      for (int c = lane; c < (K_NBR * DIM) / 4; c += 32)
        ((float4*)&L.xs[w][0])[c] = xf4[c];
#endif
    } else {
      float4 z; z.x = z.y = z.z = z.w = 0.f;
      for (int c = lane; c < (K_NBR * DIM) / 4; c += 32)
        ((float4*)&L.xs[w][0])[c] = z;
    }
    __syncthreads();

    // ---- xe = x @ W_emb^T : 8 N-tiles x 4 K-tiles of v_wmma_f32_16x16x32_bf16 ----
    v8f acc[8];
#pragma unroll
    for (int j = 0; j < 8; ++j) acc[j] = (v8f)(0.f);
#pragma unroll
    for (int t = 0; t < 4; ++t) {
      const int kbA = 32 * t + (half ? 8 : 0);       // A layout: {0..7,16..23} | {8..15,24..31}
      ABFrag a;                                      // convert f32 x -> bf16 in-register
      a.h[0] = cvt8(&L.xs[w][l16 * DIM + kbA]);
      a.h[1] = cvt8(&L.xs[w][l16 * DIM + kbA + 16]);
      const int kbB = 32 * t + (half ? 8 : 0);       // B layout: {0..7,16..23} | {8..15,24..31}
#pragma unroll
      for (int j = 0; j < 8; ++j) {
        const int ncol = j * 16 + l16;
        ABFrag b;
        b.h[0] = *(const v8bf*)&L.Wemb[ncol * DIM + kbB];
        b.h[1] = *(const v8bf*)&L.Wemb[ncol * DIM + kbB + 16];
        acc[j] = wmma_bf16(a, b, acc[j]);
      }
    }
    __syncthreads();   // all lanes done reading x from L.xs; safe to overwrite with xe

    // ---- spill xe as f32 (weighted sum) and bf16 (score WMMA A) ----
#pragma unroll
    for (int j = 0; j < 8; ++j) {
#pragma unroll
      for (int r = 0; r < 8; ++r) {
        const int m = r + 8 * half;
        const int d = j * 16 + l16;
        const float v = acc[j][r];
        L.xs  [w][m * DIM + d] = v;
        L.xebf[w][m * DIM + d] = f2bf(v);
      }
    }
    __syncthreads();

    // ---- scores: one 16x16 tile = xe_bf16 @ [W_nbr; W_self]^T (4 WMMA) ----
    v8f sacc = (v8f)(0.f);
#pragma unroll
    for (int t = 0; t < 4; ++t) {
      const int kbA = 32 * t + (half ? 8 : 0);
      ABFrag a;
      a.h[0] = *(const v8bf*)&L.xebf[w][l16 * DIM + kbA];
      a.h[1] = *(const v8bf*)&L.xebf[w][l16 * DIM + kbA + 16];
      const int kbB = 32 * t + (half ? 8 : 0);
      ABFrag b;
      b.h[0] = *(const v8bf*)&L.Wsc[l16 * DIM + kbB];
      b.h[1] = *(const v8bf*)&L.Wsc[l16 * DIM + kbB + 16];
      sacc = wmma_bf16(a, b, sacc);
    }
#pragma unroll
    for (int r = 0; r < 8; ++r)
      L.sc[w][(r + 8 * half) * 16 + l16] = sacc[r];
    __syncthreads();

    // ---- leaky-relu + mask + softmax over k (one lane per head) ----
    if (lane < NHEAD) {
      const int h = lane;
      const float selfterm = L.sc[w][0 * 16 + (8 + h)];
      float s[K_NBR];
      float mx = -__builtin_inff();
#pragma unroll
      for (int k = 0; k < K_NBR; ++k) {
        float raw = L.sc[w][k * 16 + h] + selfterm;
        float lr  = raw > 0.f ? raw : NEG_SLOPE * raw;
        bool  mk  = valid ? (mask[(size_t)n * K_NBR + k] != 0) : (k == 0);
        s[k] = mk ? lr : -__builtin_inff();
        mx = fmaxf(mx, s[k]);
      }
      float sum = 0.f;
#pragma unroll
      for (int k = 0; k < K_NBR; ++k) {
        float e = (s[k] == -__builtin_inff()) ? 0.f : __expf(s[k] - mx);
        s[k] = e; sum += e;
      }
      const float inv = 1.f / sum;
#pragma unroll
      for (int k = 0; k < K_NBR; ++k) L.pp[w][k * NHEAD + h] = s[k] * inv;
    }
    __syncthreads();

    // ---- per-head weighted sum: out[d] = sum_k p[k,h(d)] * xe[k,d] ----
#pragma unroll
    for (int q = 0; q < 4; ++q) {
      const int d = lane + 32 * q;
      const int h = d >> 4;
      float a = 0.f;
#pragma unroll
      for (int k = 0; k < K_NBR; ++k)
        a = fmaf(L.pp[w][k * NHEAD + h], L.xs[w][k * DIM + d], a);
      L.row[w][i * DIM + d] = f2bf(valid ? a : 0.f);
    }
    __syncthreads();
  }

  // ---- final: Y[16 items x 128] = row @ W_fin^T (8 N-tiles x 4 K-tiles WMMA) ----
  v8f oacc[8];
#pragma unroll
  for (int j = 0; j < 8; ++j) oacc[j] = (v8f)(0.f);
#pragma unroll
  for (int t = 0; t < 4; ++t) {
    const int kbA = 32 * t + (half ? 8 : 0);
    ABFrag a;
    a.h[0] = *(const v8bf*)&L.row[w][l16 * DIM + kbA];
    a.h[1] = *(const v8bf*)&L.row[w][l16 * DIM + kbA + 16];
    const int kbB = 32 * t + (half ? 8 : 0);
#pragma unroll
    for (int j = 0; j < 8; ++j) {
      const int ocol = j * 16 + l16;
      ABFrag b;
      b.h[0] = *(const v8bf*)&L.Wfin[ocol * DIM + kbB];
      b.h[1] = *(const v8bf*)&L.Wfin[ocol * DIM + kbB + 16];
      oacc[j] = wmma_bf16(a, b, oacc[j]);
    }
  }
#pragma unroll
  for (int j = 0; j < 8; ++j) {
#pragma unroll
    for (int r = 0; r < 8; ++r) {
      const int m  = r + 8 * half;           // item index within wave group
      const int n2 = waveBase + m;
      if (n2 < N) out[(size_t)n2 * DIM + j * 16 + l16] = oacc[j][r];
    }
  }
}

extern "C" void kernel_launch(void* const* d_in, const int* in_sizes, int n_in,
                              void* d_out, int out_size, void* d_ws, size_t ws_size,
                              hipStream_t stream) {
  const float*         x      = (const float*)d_in[0];
  const unsigned char* mask   = (const unsigned char*)d_in[1];
  const float*         Wemb   = (const float*)d_in[2];
  const float*         Wscore = (const float*)d_in[3];
  const float*         Wfin   = (const float*)d_in[4];
  float*               out    = (float*)d_out;

  const int N = in_sizes[0] / (K_NBR * DIM);
  const int grid = (N + ITEMS_PER_BLOCK - 1) / ITEMS_PER_BLOCK;
  gca_kernel<<<dim3(grid), dim3(THREADS), 0, stream>>>(x, mask, Wemb, Wscore, Wfin, out, N);
}
